// DGCNNSegmentation_70695161692408
// MI455X (gfx1250) — compile-verified, matrix operand order checked by reading, not run on hardware
//
#include <hip/hip_runtime.h>
#include <hip/hip_bf16.h>

typedef __attribute__((ext_vector_type(16))) _Float16 v16h;
typedef __attribute__((ext_vector_type(8)))  _Float16 v8h;
typedef __attribute__((ext_vector_type(8)))  float    v8f;

#define NEG_INF (-3.402823466e38f)
#define KNN_K 20
#define KNN_WAVES 4

__device__ __forceinline__ void topk_insert(float (&kv)[KNN_K], int (&ki)[KNN_K],
                                            float val, int idx) {
    float cv = val; int ci = idx;
    #pragma unroll
    for (int p = 0; p < KNN_K; ++p) {
        bool sw = cv > kv[p];
        float tv = kv[p]; int ti = ki[p];
        kv[p] = sw ? cv : tv;  ki[p] = sw ? ci : ti;
        cv    = sw ? tv : cv;  ci    = sw ? ti : ci;
    }
}

// ---------------------------------------------------------------------------
// Pack features (column-major per batch: in[(b*c+ci)*quot + q]) into f16
// rows padded to 64, and compute squared norms.
// ---------------------------------------------------------------------------
__global__ void pack_norm_kernel(const float* __restrict__ in,
                                 _Float16* __restrict__ x16,
                                 float* __restrict__ norms,
                                 int n, int c, int quot) {
    int i = blockIdx.x * blockDim.x + threadIdx.x;
    if (i >= n) return;
    int b = i / quot;
    int q = i - b * quot;
    float s = 0.0f;
    #pragma unroll
    for (int ci = 0; ci < 64; ++ci) {
        float v = (ci < c) ? in[(size_t)(b * c + ci) * quot + q] : 0.0f;
        s += v * v;
        x16[(size_t)i * 64 + ci] = (_Float16)v;
    }
    norms[i] = s;
}

// ---------------------------------------------------------------------------
// Fused KNN: 4 waves per block; each wave owns the same 16-row tile but a
// disjoint quarter of the columns. Per 16-col tile: Gram via 2x WMMA f16
// (K=64), pd epilogue, LDS transpose (wave-private region, no block barrier),
// float4 scan into per-lane register top-20. Final cross-wave merge of the
// 4x20 candidate lists per row through LDS.
// ---------------------------------------------------------------------------
__global__ __launch_bounds__(32 * KNN_WAVES)
void knn_kernel(const _Float16* __restrict__ x16,
                const float* __restrict__ norms,
                int* __restrict__ knn,
                int quot) {
    int tid   = threadIdx.x;
    int lane  = tid & 31;
    int wid   = tid >> 5;
    int mt    = blockIdx.x;          // row tile within batch
    int b     = blockIdx.y;          // batch
    int lo16  = lane & 15;
    int khalf = lane >> 4;

    int rowpt = b * quot + mt * 16 + lo16;

    // A tile (16 rows x K64) in CDNA5 16-bit A striping.
    const v8h* arow = (const v8h*)(x16 + (size_t)rowpt * 64);
    v8h a0 = arow[0 + khalf];
    v8h a1 = arow[2 + khalf];
    v8h a2 = arow[4 + khalf];
    v8h a3 = arow[6 + khalf];
    v16h alo = __builtin_shufflevector(a0, a1, 0,1,2,3,4,5,6,7,8,9,10,11,12,13,14,15);
    v16h ahi = __builtin_shufflevector(a2, a3, 0,1,2,3,4,5,6,7,8,9,10,11,12,13,14,15);

    __shared__ float srn[16];
    __shared__ float spd[KNN_WAVES * 256];
    __shared__ float lval[KNN_WAVES * 16 * KNN_K];
    __shared__ int   lidx[KNN_WAVES * 16 * KNN_K];

    if (tid < 16) srn[tid] = norms[b * quot + mt * 16 + tid];
    __syncthreads();
    float rn[8];
    #pragma unroll
    for (int v = 0; v < 8; ++v) rn[v] = srn[v + 8 * khalf];

    float kv[KNN_K];
    int   ki[KNN_K];
    #pragma unroll
    for (int t = 0; t < KNN_K; ++t) { kv[t] = NEG_INF; ki[t] = 0; }
    float kmin = NEG_INF;

    int seg    = quot / KNN_WAVES;   // columns per wave (2048)
    int cbase  = wid * seg;
    int ltiles = seg >> 4;           // 128
    float* myspd = spd + wid * 256;

    // Software-pipelined B loads (register double buffer).
    int colpt0 = b * quot + cbase + lo16;
    const v8h* brow0 = (const v8h*)(x16 + (size_t)colpt0 * 64);
    v8h nb0 = brow0[0 + khalf];
    v8h nb1 = brow0[2 + khalf];
    v8h nb2 = brow0[4 + khalf];
    v8h nb3 = brow0[6 + khalf];
    float ncn = norms[colpt0];

    for (int jt = 0; jt < ltiles; ++jt) {
        v8h b0 = nb0, b1 = nb1, b2 = nb2, b3 = nb3;
        float cn = ncn;

        // Issue next tile's loads before compute/scan (latency hiding).
        int jn  = (jt + 1 < ltiles) ? jt + 1 : jt;
        int cpt = b * quot + cbase + jn * 16 + lo16;
        const v8h* nbrow = (const v8h*)(x16 + (size_t)cpt * 64);
        nb0 = nbrow[0 + khalf];
        nb1 = nbrow[2 + khalf];
        nb2 = nbrow[4 + khalf];
        nb3 = nbrow[6 + khalf];
        ncn = norms[cpt];
        if (jt + 8 < ltiles)
            __builtin_prefetch(x16 + (size_t)(b * quot + cbase + (jt + 8) * 16 + lo16) * 64, 0, 1);

        v16h blo = __builtin_shufflevector(b0, b1, 0,1,2,3,4,5,6,7,8,9,10,11,12,13,14,15);
        v16h bhi = __builtin_shufflevector(b2, b3, 0,1,2,3,4,5,6,7,8,9,10,11,12,13,14,15);

        v8f acc = {};
        acc = __builtin_amdgcn_wmma_f32_16x16x32_f16(false, alo, false, blo,
                                                     (short)0, acc, false, false);
        acc = __builtin_amdgcn_wmma_f32_16x16x32_f16(false, ahi, false, bhi,
                                                     (short)0, acc, false, false);

        // pd[m][n] = 2*S - |r|^2 - |c|^2 ; transpose via wave-private LDS
        #pragma unroll
        for (int v = 0; v < 8; ++v) {
            int m = v + 8 * khalf;
            myspd[m * 16 + lo16] = 2.0f * acc[v] - rn[v] - cn;
        }
        __builtin_amdgcn_wave_barrier();
        asm volatile("" ::: "memory");

        if (lane < 16) {
            const float4* rowp = (const float4*)(myspd + lane * 16);
            int base = cbase + jt * 16;
            #pragma unroll
            for (int t4 = 0; t4 < 4; ++t4) {
                float4 vv = rowp[t4];
                float mx = fmaxf(fmaxf(vv.x, vv.y), fmaxf(vv.z, vv.w));
                if (mx > kmin) {
                    int bi = base + t4 * 4;
                    if (vv.x > kmin) { topk_insert(kv, ki, vv.x, bi + 0); kmin = kv[KNN_K-1]; }
                    if (vv.y > kmin) { topk_insert(kv, ki, vv.y, bi + 1); kmin = kv[KNN_K-1]; }
                    if (vv.z > kmin) { topk_insert(kv, ki, vv.z, bi + 2); kmin = kv[KNN_K-1]; }
                    if (vv.w > kmin) { topk_insert(kv, ki, vv.w, bi + 3); kmin = kv[KNN_K-1]; }
                }
            }
        }
        __builtin_amdgcn_wave_barrier();
        asm volatile("" ::: "memory");
    }

    // Dump per-wave lists, then wave 0 merges 4x20 candidates per row.
    if (lane < 16) {
        int lb = (wid * 16 + lane) * KNN_K;
        #pragma unroll
        for (int t = 0; t < KNN_K; ++t) { lval[lb + t] = kv[t]; lidx[lb + t] = ki[t]; }
    }
    __syncthreads();

    if (tid < 16) {
        float fv[KNN_K];
        int   fi[KNN_K];
        #pragma unroll
        for (int t = 0; t < KNN_K; ++t) { fv[t] = NEG_INF; fi[t] = 0; }
        float fmin = NEG_INF;
        for (int w = 0; w < KNN_WAVES; ++w) {
            int lb = (w * 16 + tid) * KNN_K;
            #pragma unroll
            for (int t = 0; t < KNN_K; ++t) {
                float val = lval[lb + t];
                if (val > fmin) { topk_insert(fv, fi, val, lidx[lb + t]); fmin = fv[KNN_K-1]; }
            }
        }
        int* o = knn + (size_t)(b * quot + mt * 16 + tid) * KNN_K;
        #pragma unroll
        for (int t = 0; t < KNN_K; ++t) o[t] = fi[t];
    }
}

// ---------------------------------------------------------------------------
// Edge feature + max-pool, using max_k([nbr-ctr, ctr]) = [max_k(nbr)-ctr, ctr]
// ---------------------------------------------------------------------------
__global__ void graphfeat_kernel(const float* __restrict__ in,
                                 const int* __restrict__ knn,
                                 float* __restrict__ f,
                                 int n, int c, int quot) {
    int t = blockIdx.x * blockDim.x + threadIdx.x;
    if (t >= n * c) return;
    int i = t / c;
    int d = t - i * c;
    int b = i / quot;
    int q = i - b * quot;
    const float* ch = in + (size_t)(b * c + d) * quot;
    float ctr = ch[q];
    const int* ids = knn + (size_t)i * KNN_K;
    float best = NEG_INF;
    #pragma unroll
    for (int kk = 0; kk < KNN_K; ++kk) best = fmaxf(best, ch[ids[kk]]);
    int F = 2 * c;
    f[(size_t)i * F + d]     = best - ctr;
    f[(size_t)i * F + c + d] = ctr;
}

// ---------------------------------------------------------------------------
// Utility kernels
// ---------------------------------------------------------------------------
__global__ void zero_kernel(float* __restrict__ p, unsigned int count) {
    unsigned int i = blockIdx.x * blockDim.x + threadIdx.x;
    if (i < count) p[i] = 0.0f;
}

__global__ void degree_kernel(const int* __restrict__ src, const int* __restrict__ dst,
                              float* __restrict__ degO, float* __restrict__ degI, int e) {
    int i = blockIdx.x * blockDim.x + threadIdx.x;
    if (i >= e) return;
    atomicAdd(&degO[src[i]], 1.0f);
    atomicAdd(&degI[dst[i]], 1.0f);
}

__global__ void invnorm_kernel(float* __restrict__ d, int n) {
    int i = blockIdx.x * blockDim.x + threadIdx.x;
    if (i >= n) return;
    float v = d[i];
    d[i] = (v > 0.0f) ? rsqrtf(fmaxf(v, 1.0f)) : 0.0f;
}

__global__ void scatter_kernel(const float* __restrict__ f,
                               const int* __restrict__ src, const int* __restrict__ dst,
                               const float* __restrict__ ns,
                               float* __restrict__ agg, int F, unsigned int total) {
    unsigned int t = blockIdx.x * blockDim.x + threadIdx.x;
    if (t >= total) return;
    int e = t / (unsigned int)F;
    int d = t - e * (unsigned int)F;
    int s = src[e];
    int ds = dst[e];
    atomicAdd(&agg[(size_t)ds * F + d], f[(size_t)s * F + d] * ns[s]);
}

__global__ void scaleconv_kernel(const float* __restrict__ agg,
                                 const float* __restrict__ nd,
                                 _Float16* __restrict__ g16,
                                 int n, int F, int Kpad) {
    unsigned int t = blockIdx.x * blockDim.x + threadIdx.x;
    if (t >= (unsigned int)n * (unsigned int)Kpad) return;
    int i = t / (unsigned int)Kpad;
    int d = t - i * (unsigned int)Kpad;
    float v = (d < F) ? agg[(size_t)i * F + d] * nd[i] : 0.0f;
    g16[t] = (_Float16)v;
}

// ---------------------------------------------------------------------------
// WMMA GEMM: out[m][n] = sum_k g16[m][k] * W[k][n] + bias[n], optional ReLU.
// ---------------------------------------------------------------------------
__global__ __launch_bounds__(32) void gemm_kernel(const _Float16* __restrict__ g16,
                                                  const float* __restrict__ W,
                                                  const float* __restrict__ bias,
                                                  float* __restrict__ out,
                                                  int Kpad, int Kreal, int Nreal,
                                                  int relu) {
    int lane  = threadIdx.x;
    int mt    = blockIdx.x;
    int nt    = blockIdx.y;
    int lo16  = lane & 15;
    int khalf = lane >> 4;
    int mrow  = mt * 16 + lo16;
    int ncol  = nt * 16 + lo16;

    v8f acc = {};
    int ksteps = Kpad >> 5;
    for (int ks = 0; ks < ksteps; ++ks) {
        const v8h* arow = (const v8h*)(g16 + (size_t)mrow * Kpad + ks * 32);
        v8h A0 = arow[0 + khalf];
        v8h A1 = arow[2 + khalf];
        v16h av = __builtin_shufflevector(A0, A1, 0,1,2,3,4,5,6,7,8,9,10,11,12,13,14,15);

        v16h bv;
        #pragma unroll
        for (int i = 0; i < 8; ++i) {
            int k1 = ks * 32 + 8 * khalf + i;
            int k2 = ks * 32 + 16 + 8 * khalf + i;
            float w1 = (k1 < Kreal && ncol < Nreal) ? W[(size_t)k1 * Nreal + ncol] : 0.0f;
            float w2 = (k2 < Kreal && ncol < Nreal) ? W[(size_t)k2 * Nreal + ncol] : 0.0f;
            bv[i]     = (_Float16)w1;
            bv[8 + i] = (_Float16)w2;
        }
        acc = __builtin_amdgcn_wmma_f32_16x16x32_f16(false, av, false, bv,
                                                     (short)0, acc, false, false);
    }

    float bb = (ncol < Nreal) ? bias[ncol] : 0.0f;
    #pragma unroll
    for (int v = 0; v < 8; ++v) {
        int m = mt * 16 + v + 8 * khalf;
        float val = acc[v] + bb;
        if (relu) val = fmaxf(val, 0.0f);
        if (ncol < Nreal) out[(size_t)m * Nreal + ncol] = val;
    }
}

// ---------------------------------------------------------------------------
// Host-side orchestration
// ---------------------------------------------------------------------------
extern "C" void kernel_launch(void* const* d_in, const int* in_sizes, int n_in,
                              void* d_out, int out_size, void* d_ws, size_t ws_size,
                              hipStream_t stream) {
    (void)n_in; (void)out_size; (void)ws_size;
    const float* feat = (const float*)d_in[0];
    const int*   src  = (const int*)d_in[1];
    const int*   dst  = (const int*)d_in[2];
    const float* W1   = (const float*)d_in[3];
    const float* b1   = (const float*)d_in[4];
    const float* W2   = (const float*)d_in[5];
    const float* b2   = (const float*)d_in[6];
    const float* W3   = (const float*)d_in[7];
    const float* b3   = (const float*)d_in[8];

    const int n    = in_sizes[0] / 3;   // 32768
    const int e    = in_sizes[1];       // 524288
    const int quot = n / 4;             // 8192

    char* ws = (char*)d_ws;
    size_t off = 0;
    auto salloc = [&](size_t bytes) -> char* {
        char* p = ws + off;
        off = (off + bytes + 255) & ~(size_t)255;
        return p;
    };
    _Float16* x16   = (_Float16*)salloc((size_t)n * 64 * sizeof(_Float16));
    float*    norms = (float*)   salloc((size_t)n * sizeof(float));
    int*      idx   = (int*)     salloc((size_t)n * KNN_K * sizeof(int));
    float*    hA    = (float*)   salloc((size_t)n * 64 * sizeof(float));
    float*    hB    = (float*)   salloc((size_t)n * 64 * sizeof(float));
    float*    fbuf  = (float*)   salloc((size_t)n * 128 * sizeof(float));
    float*    ns    = (float*)   salloc((size_t)n * sizeof(float));
    float*    nd    = (float*)   salloc((size_t)n * sizeof(float));
    float*    agg   = (float*)   salloc((size_t)n * 128 * sizeof(float));
    _Float16* g16   = (_Float16*)salloc((size_t)n * 128 * sizeof(_Float16));

    const int B256 = 256;
    auto blk = [](unsigned int total, int b) { return (total + b - 1) / b; };

    // Degrees / normalizers once (identical across layers).
    zero_kernel<<<blk(n, B256), B256, 0, stream>>>(ns, n);
    zero_kernel<<<blk(n, B256), B256, 0, stream>>>(nd, n);
    degree_kernel<<<blk(e, B256), B256, 0, stream>>>(src, dst, ns, nd, e);
    invnorm_kernel<<<blk(n, B256), B256, 0, stream>>>(ns, n);
    invnorm_kernel<<<blk(n, B256), B256, 0, stream>>>(nd, n);

    auto layer = [&](const float* hin, int c, const float* W, const float* bias,
                     int Nout, int relu, float* hout) {
        int F    = 2 * c;
        int Kpad = (F <= 32) ? 32 : 128;
        pack_norm_kernel<<<blk(n, B256), B256, 0, stream>>>(hin, x16, norms, n, c, quot);
        knn_kernel<<<dim3(quot / 16, 4), 32 * KNN_WAVES, 0, stream>>>(x16, norms, idx, quot);
        graphfeat_kernel<<<blk((unsigned int)n * c, B256), B256, 0, stream>>>(
            hin, idx, fbuf, n, c, quot);
        zero_kernel<<<blk((unsigned int)n * F, B256), B256, 0, stream>>>(agg, (unsigned int)n * F);
        unsigned int total = (unsigned int)e * (unsigned int)F;
        scatter_kernel<<<blk(total, B256), B256, 0, stream>>>(fbuf, src, dst, ns, agg, F, total);
        scaleconv_kernel<<<blk((unsigned int)n * Kpad, B256), B256, 0, stream>>>(
            agg, nd, g16, n, F, Kpad);
        gemm_kernel<<<dim3(n / 16, (Nout + 15) / 16), 32, 0, stream>>>(
            g16, W, bias, hout, Kpad, F, Nout, relu);
    };

    layer(feat, 3,  W1, b1, 64, 1, hA);            // h1 = relu(conv(f1))
    layer(hA,   64, W2, b2, 64, 1, hB);            // h2 = relu(conv(f2))
    layer(hB,   64, W3, b3, 8,  0, (float*)d_out); // out = conv(f3)
}
